// GAT_71116068488098
// MI455X (gfx1250) — matrix operand order repaired and checked by reading, not compiled, not run on hardware
//
#include <hip/hip_runtime.h>
#include <hip/hip_bf16.h>

// ---------------- problem constants (from reference) ----------------
#define NN   50000      // nodes
#define EE   1250000    // raw edges
#define EAT  1300000    // edges + self loops (EE + NN)
#define DIN  64         // K for every layer GEMM (64 in, 64/64/32 out)
#define ED   16
#define HC   64         // H*C for layers 1,2
#define CC   32
#define GG   16
#define NCLS 5
#define SLOPE 0.2f

typedef __attribute__((ext_vector_type(16))) __bf16 v16bf;
typedef __attribute__((ext_vector_type(8)))  float  v8f;

union Frag32 {            // one WMMA 16-bit fragment: 16 bf16 = 32 bytes
  uint4 q[2];
  v16bf v;
};

static inline int cdiv_i(int a, int b) { return (a + b - 1) / b; }

// ---------------- small helpers ----------------
__device__ __forceinline__ void atomicMaxF(float* addr, float val) {
  if (val >= 0.0f) atomicMax((int*)addr, __float_as_int(val));
  else             atomicMin((unsigned int*)addr, (unsigned int)__float_as_int(val));
}

__device__ __forceinline__ void edge_sd(int e, const int* __restrict__ ei, int& s, int& d) {
  if (e < EE) { s = ei[e]; d = ei[EE + e]; }
  else        { s = e - EE; d = e - EE; }      // appended self loop
}

__global__ void k_fill_f32(float* __restrict__ p, float v, int n) {
  int t = blockIdx.x * blockDim.x + threadIdx.x;
  if (t < n) p[t] = v;
}

// ---------------- self-loop attr = mean of incoming edge_attr ----------------
__global__ void k_deg_attr(const int* __restrict__ ei, const float* __restrict__ eattr,
                           float* __restrict__ deg, float* __restrict__ selfat) {
  int e = blockIdx.x * blockDim.x + threadIdx.x;
  if (e >= EE) return;
  int d = ei[EE + e];
  atomicAdd(&deg[d], 1.0f);
  const float4* ea = (const float4*)(eattr + (size_t)e * ED);
  float* sa = selfat + (size_t)d * ED;
  #pragma unroll
  for (int q = 0; q < 4; ++q) {
    float4 v = ea[q];
    atomicAdd(&sa[q * 4 + 0], v.x);
    atomicAdd(&sa[q * 4 + 1], v.y);
    atomicAdd(&sa[q * 4 + 2], v.z);
    atomicAdd(&sa[q * 4 + 3], v.w);
  }
}

__global__ void k_selfattr_norm(float* __restrict__ selfat, const float* __restrict__ deg) {
  int t = blockIdx.x * blockDim.x + threadIdx.x;
  if (t >= NN * ED) return;
  selfat[t] /= fmaxf(deg[t >> 4], 1.0f);
}

// ---------------- precontract We with ae: wea[d][h] = sum_c We[d, h*C+c] * ae[h,c] ----
// wea layout: [0..31] layer1 (d*2+h), [32..63] layer2 (d*2+h), [64..79] layer3 (d)
__global__ void k_wea(const float* __restrict__ We1, const float* __restrict__ ae1,
                      const float* __restrict__ We2, const float* __restrict__ ae2,
                      const float* __restrict__ We3, const float* __restrict__ ae3,
                      float* __restrict__ wea) {
  int t = threadIdx.x;
  if (t < 32) {
    int d = t >> 1, h = t & 1; float s = 0.f;
    for (int c = 0; c < CC; ++c) s += We1[d * HC + h * CC + c] * ae1[h * CC + c];
    wea[t] = s;
  } else if (t < 64) {
    int u = t - 32; int d = u >> 1, h = u & 1; float s = 0.f;
    for (int c = 0; c < CC; ++c) s += We2[d * HC + h * CC + c] * ae2[h * CC + c];
    wea[t] = s;
  } else if (t < 80) {
    int d = t - 64; float s = 0.f;
    for (int c = 0; c < CC; ++c) s += We3[d * CC + c] * ae3[c];
    wea[t] = s;
  }
}

// ---------------- per-edge scores s_e for all 3 layers in one streaming pass ----------
__global__ void k_se(const float* __restrict__ eattr, const float* __restrict__ selfat,
                     const float* __restrict__ wea,
                     float* __restrict__ se1, float* __restrict__ se2, float* __restrict__ se3) {
  int e = blockIdx.x * blockDim.x + threadIdx.x;
  if (e >= EAT) return;
  const float* ap = (e < EE) ? (eattr + (size_t)e * ED) : (selfat + (size_t)(e - EE) * ED);
  __builtin_prefetch(ap, 0, 0);
  float a[ED];
  #pragma unroll
  for (int d = 0; d < ED; ++d) a[d] = ap[d];
  float s10 = 0.f, s11 = 0.f, s20 = 0.f, s21 = 0.f, s30 = 0.f;
  #pragma unroll
  for (int d = 0; d < ED; ++d) {
    s10 += a[d] * wea[d * 2 + 0];
    s11 += a[d] * wea[d * 2 + 1];
    s20 += a[d] * wea[32 + d * 2 + 0];
    s21 += a[d] * wea[32 + d * 2 + 1];
    s30 += a[d] * wea[64 + d];
  }
  se1[e * 2 + 0] = s10; se1[e * 2 + 1] = s11;
  se2[e * 2 + 0] = s20; se2[e * 2 + 1] = s21;
  se3[e] = s30;
}

// ---------------- f32 -> bf16 conversion (activations) ----------------
__global__ void k_cvt_bf16(const float* __restrict__ src, __bf16* __restrict__ dst, int n) {
  int t = blockIdx.x * blockDim.x + threadIdx.x;
  if (t < n) dst[t] = (__bf16)src[t];
}

// ---------------- weight swizzle: f32 row-major [K x Ncol] -> bf16 fragment-major -----
// Bswz[((kt*nct + ct)*32 + lane)*16 + i] holds W[kt*32 + 16*(i>>3) + 8*(lane>>4) + (i&7)][ct*16 + (lane&15)]
// so each lane's 16-element B fragment is a contiguous, 32B-aligned run.
__global__ void k_swz_b(const float* __restrict__ W, __bf16* __restrict__ Bswz,
                        int K, int Ncol) {
  int t = blockIdx.x * blockDim.x + threadIdx.x;
  if (t >= K * Ncol) return;
  int i    = t & 15;
  int lane = (t >> 4) & 31;
  int tile = t >> 9;                 // kt * nct + ct
  int nct  = Ncol >> 4;
  int kt   = tile / nct, ct = tile % nct;
  int k    = kt * 32 + ((i >> 3) << 4) + ((lane >> 4) << 3) + (i & 7);
  int col  = ct * 16 + (lane & 15);
  Bswz[t] = (__bf16)W[(size_t)k * Ncol + col];
}

// ---------------- WMMA GEMM: C[M x NCOL] = A[M x 64] * B[64 x NCOL] ----------------
// A: bf16 row-major (K = DIN = 64 fixed); Bswz: bf16 fragment-major (k_swz_b);
// C: f32 row-major. One wave per 16-row tile; M % 16 == 0 (50000 = 3125*16).
// NCOL is a compile-time constant so all D-store offsets are immediates.
template <int NCOL>
__global__ void k_gemm_bf16_wmma(const __bf16* __restrict__ A, const __bf16* __restrict__ Bswz,
                                 float* __restrict__ C, int M) {
  constexpr int NCT = NCOL / 16;
  int wave = (blockIdx.x * blockDim.x + threadIdx.x) >> 5;
  int lane = threadIdx.x & 31;
  if (wave >= (M >> 4)) return;           // wave-uniform: EXEC all-ones for WMMA
  int row0     = wave << 4;
  int laneHalf = lane >> 4;               // 0 | 1
  int lr       = lane & 15;
  int kbase    = laneHalf * 8;

  // Preload both A fragments (K=64 -> kt=0,1); 16B-aligned vector loads.
  const __bf16* ap = A + (size_t)(row0 + lr) * DIN + kbase;
  Frag32 a0, a1;
  a0.q[0] = *(const uint4*)(ap);       a0.q[1] = *(const uint4*)(ap + 16);
  a1.q[0] = *(const uint4*)(ap + 32);  a1.q[1] = *(const uint4*)(ap + 48);

  // One output base pointer; every store offset below is a compile-time immediate.
  float* crow = C + (size_t)(row0 + laneHalf * 8) * NCOL + lr;

  #pragma unroll
  for (int ct = 0; ct < NCT; ++ct) {
    const __bf16* bp0 = Bswz + ((size_t)(0 * NCT + ct) * 32 + lane) * 16;
    const __bf16* bp1 = Bswz + ((size_t)(1 * NCT + ct) * 32 + lane) * 16;
    Frag32 b0, b1;
    b0.q[0] = *(const uint4*)(bp0); b0.q[1] = *(const uint4*)(bp0 + 8);
    b1.q[0] = *(const uint4*)(bp1); b1.q[1] = *(const uint4*)(bp1 + 8);
    v8f acc = {};
    acc = __builtin_amdgcn_wmma_f32_16x16x32_bf16(false, a0.v, false, b0.v,
                                                  (short)0, acc, false, false);
    acc = __builtin_amdgcn_wmma_f32_16x16x32_bf16(false, a1.v, false, b1.v,
                                                  (short)0, acc, false, false);
    #pragma unroll
    for (int j = 0; j < 8; ++j)
      crow[j * NCOL + ct * 16] = acc[j];   // immediate offsets -> clause of stores
  }
}

// ---------------- per-node attention scores ----------------
__global__ void k_scores(const float* __restrict__ h, const float* __restrict__ asrc,
                         const float* __restrict__ adst,
                         float* __restrict__ ssrc, float* __restrict__ sdst, int H) {
  int t = blockIdx.x * blockDim.x + threadIdx.x;
  if (t >= NN * H) return;
  int n = t / H, hh = t % H;
  const float* hp = h + (size_t)n * (H * CC) + hh * CC;
  float a = 0.f, b = 0.f;
  #pragma unroll
  for (int c = 0; c < CC; ++c) { a += hp[c] * asrc[hh * CC + c]; b += hp[c] * adst[hh * CC + c]; }
  ssrc[t] = a; sdst[t] = b;
}

// ---------------- edge pass 1: leaky-relu logits + segment max ----------------
__global__ void k_logit_max(const int* __restrict__ ei, const float* __restrict__ ssrc,
                            const float* __restrict__ sdst, const float* __restrict__ se,
                            float* __restrict__ logit, float* __restrict__ m, int H) {
  int t = blockIdx.x * blockDim.x + threadIdx.x;
  if (t >= EAT * H) return;
  int e = t / H, hh = t % H, s, d;
  edge_sd(e, ei, s, d);
  float v = ssrc[s * H + hh] + sdst[d * H + hh] + se[e * H + hh];
  v = (v > 0.f) ? v : SLOPE * v;
  logit[t] = v;
  atomicMaxF(&m[d * H + hh], v);
}

// ---------------- edge pass 2: exp + segment sum ----------------
__global__ void k_exp_sum(const int* __restrict__ ei, float* __restrict__ logit,
                          const float* __restrict__ m, float* __restrict__ den, int H) {
  int t = blockIdx.x * blockDim.x + threadIdx.x;
  if (t >= EAT * H) return;
  int e = t / H, hh = t % H, s, d;
  edge_sd(e, ei, s, d);
  float ex = __expf(logit[t] - m[d * H + hh]);
  logit[t] = ex;
  atomicAdd(&den[d * H + hh], ex);
}

// ---------------- edge pass 3: weighted scatter-aggregate ----------------
__global__ void k_agg(const int* __restrict__ ei, const float* __restrict__ logit,
                      const float* __restrict__ den, const float* __restrict__ h,
                      float* __restrict__ agg, int H) {
  int t = blockIdx.x * blockDim.x + threadIdx.x;
  if (t >= EAT * H) return;
  int e = t / H, hh = t % H, s, d;
  edge_sd(e, ei, s, d);
  const float* hs = h + (size_t)s * (H * CC) + hh * CC;
  __builtin_prefetch(hs, 0, 0);                       // gfx1250 global_prefetch on gather row
  float alpha = logit[t] / (den[d * H + hh] + 1e-16f);
  float* ag = agg + (size_t)d * (H * CC) + hh * CC;
  const float4* h4 = (const float4*)hs;
  #pragma unroll
  for (int q = 0; q < CC / 4; ++q) {
    float4 v = h4[q];
    atomicAdd(&ag[q * 4 + 0], v.x * alpha);
    atomicAdd(&ag[q * 4 + 1], v.y * alpha);
    atomicAdd(&ag[q * 4 + 2], v.z * alpha);
    atomicAdd(&ag[q * 4 + 3], v.w * alpha);
  }
}

// ---------------- epilogues ----------------
// layers 1-2: relu(agg + b) -> bf16 input of next layer
__global__ void k_epilogue_bf(const float* __restrict__ agg, const float* __restrict__ b,
                              __bf16* __restrict__ xbf) {
  int t = blockIdx.x * blockDim.x + threadIdx.x;
  if (t >= NN * HC) return;
  xbf[t] = (__bf16)fmaxf(agg[t] + b[t % HC], 0.0f);
}
// layer 3: relu(agg + b) in place (f32, [N,32])
__global__ void k_epilogue3(float* __restrict__ agg, const float* __restrict__ b) {
  int t = blockIdx.x * blockDim.x + threadIdx.x;
  if (t >= NN * CC) return;
  agg[t] = fmaxf(agg[t] + b[t % CC], 0.0f);
}

// ---------------- pooling + ffn + softmax ----------------
__global__ void k_pool(const float* __restrict__ h3, const int* __restrict__ batch,
                       float* __restrict__ pooled, float* __restrict__ cnt) {
  int t = blockIdx.x * blockDim.x + threadIdx.x;
  if (t >= NN * CC) return;
  int n = t / CC, c = t % CC, g = batch[n];
  atomicAdd(&pooled[g * CC + c], h3[t]);
  if (c == 0) atomicAdd(&cnt[g], 1.0f);
}

__global__ void k_final(const float* __restrict__ pooled, const float* __restrict__ cnt,
                        const float* __restrict__ Wf, const float* __restrict__ bf,
                        float* __restrict__ out) {
  int g = threadIdx.x;
  if (g >= GG) return;
  float inv = 1.0f / fmaxf(cnt[g], 1.0f);
  float lg[NCLS];
  #pragma unroll
  for (int j = 0; j < NCLS; ++j) {
    float s = bf[j];
    for (int c = 0; c < CC; ++c) s += pooled[g * CC + c] * inv * Wf[c * NCLS + j];
    lg[j] = s;
  }
  float mx = lg[0];
  #pragma unroll
  for (int j = 1; j < NCLS; ++j) mx = fmaxf(mx, lg[j]);
  float sum = 0.f;
  #pragma unroll
  for (int j = 0; j < NCLS; ++j) { lg[j] = __expf(lg[j] - mx); sum += lg[j]; }
  #pragma unroll
  for (int j = 0; j < NCLS; ++j) out[g * NCLS + j] = lg[j] / sum;
}

// =====================================================================
extern "C" void kernel_launch(void* const* d_in, const int* in_sizes, int n_in,
                              void* d_out, int out_size, void* d_ws, size_t ws_size,
                              hipStream_t stream) {
  const float* x     = (const float*)d_in[0];
  const int*   ei    = (const int*)  d_in[1];
  const float* eattr = (const float*)d_in[2];
  const int*   batch = (const int*)  d_in[3];
  const float* W[3]    = {(const float*)d_in[4],  (const float*)d_in[10], (const float*)d_in[16]};
  const float* asrc[3] = {(const float*)d_in[5],  (const float*)d_in[11], (const float*)d_in[17]};
  const float* adst[3] = {(const float*)d_in[6],  (const float*)d_in[12], (const float*)d_in[18]};
  const float* We[3]   = {(const float*)d_in[7],  (const float*)d_in[13], (const float*)d_in[19]};
  const float* ae[3]   = {(const float*)d_in[8],  (const float*)d_in[14], (const float*)d_in[20]};
  const float* bias[3] = {(const float*)d_in[9],  (const float*)d_in[15], (const float*)d_in[21]};
  const float* Wf = (const float*)d_in[22];
  const float* bf = (const float*)d_in[23];
  float* out = (float*)d_out;

  // ---- workspace carve-out ----
  char* p = (char*)d_ws;
  auto take = [&](size_t bytes) { char* r = p; p += (bytes + 255) & ~(size_t)255; return r; };
  float*  deg    = (float*) take((size_t)NN * 4);
  float*  selfat = (float*) take((size_t)NN * ED * 4);
  float*  wea    = (float*) take(80 * 4);
  float*  se1    = (float*) take((size_t)EAT * 2 * 4);
  float*  se2    = (float*) take((size_t)EAT * 2 * 4);
  float*  se3    = (float*) take((size_t)EAT * 4);
  __bf16* xbf    = (__bf16*)take((size_t)NN * HC * 2);
  __bf16* wswz   = (__bf16*)take((size_t)DIN * HC * 2);
  float*  h      = (float*) take((size_t)NN * HC * 4);
  float*  ssrc   = (float*) take((size_t)NN * 2 * 4);
  float*  sdst   = (float*) take((size_t)NN * 2 * 4);
  float*  m      = (float*) take((size_t)NN * 2 * 4);
  float*  den    = (float*) take((size_t)NN * 2 * 4);
  float*  logit  = (float*) take((size_t)EAT * 2 * 4);
  float*  agg    = (float*) take((size_t)NN * HC * 4);
  float*  pooled = (float*) take((size_t)GG * CC * 4);
  float*  cnt    = (float*) take((size_t)GG * 4);

  const int B = 256;

  // ---- self-loop edge_attr (mean of incoming) ----
  hipMemsetAsync(deg,    0, (size_t)NN * 4, stream);
  hipMemsetAsync(selfat, 0, (size_t)NN * ED * 4, stream);
  k_deg_attr<<<cdiv_i(EE, B), B, 0, stream>>>(ei, eattr, deg, selfat);
  k_selfattr_norm<<<cdiv_i(NN * ED, B), B, 0, stream>>>(selfat, deg);

  // ---- precontracted edge scores for all 3 layers (one pass over eattr) ----
  k_wea<<<1, 128, 0, stream>>>(We[0], ae[0], We[1], ae[1], We[2], ae[2], wea);
  k_se<<<cdiv_i(EAT, B), B, 0, stream>>>(eattr, selfat, wea, se1, se2, se3);

  // ---- input to bf16 ----
  k_cvt_bf16<<<cdiv_i(NN * DIN, B), B, 0, stream>>>(x, xbf, NN * DIN);

  const int gemmBlocks = cdiv_i(NN / 16, 8);   // 8 waves / block, 1 wave per 16-row tile
  float* seL[3] = {se1, se2, se3};

  for (int l = 0; l < 3; ++l) {
    const int H = (l == 2) ? 1 : 2;
    const int Ncol = H * CC;
    // swizzle+convert weights, then h = xbf @ W via v_wmma_f32_16x16x32_bf16
    k_swz_b<<<cdiv_i(DIN * Ncol, B), B, 0, stream>>>(W[l], wswz, DIN, Ncol);
    if (l < 2) k_gemm_bf16_wmma<64><<<gemmBlocks, B, 0, stream>>>(xbf, wswz, h, NN);
    else       k_gemm_bf16_wmma<32><<<gemmBlocks, B, 0, stream>>>(xbf, wswz, h, NN);
    // node scores
    k_scores<<<cdiv_i(NN * H, B), B, 0, stream>>>(h, asrc[l], adst[l], ssrc, sdst, H);
    // segment softmax state
    k_fill_f32<<<cdiv_i(NN * H, B), B, 0, stream>>>(m, -INFINITY, NN * H);
    hipMemsetAsync(den, 0, (size_t)NN * H * 4, stream);
    hipMemsetAsync(agg, 0, (size_t)NN * HC * 4, stream);
    // three edge passes
    k_logit_max<<<cdiv_i(EAT * H, B), B, 0, stream>>>(ei, ssrc, sdst, seL[l], logit, m, H);
    k_exp_sum  <<<cdiv_i(EAT * H, B), B, 0, stream>>>(ei, logit, m, den, H);
    k_agg      <<<cdiv_i(EAT * H, B), B, 0, stream>>>(ei, logit, den, h, agg, H);
    // epilogue
    if (l < 2) k_epilogue_bf<<<cdiv_i(NN * HC, B), B, 0, stream>>>(agg, bias[l], xbf);
    else       k_epilogue3  <<<cdiv_i(NN * CC, B), B, 0, stream>>>(agg, bias[l]);
  }

  // ---- global mean pool + ffn + softmax ----
  hipMemsetAsync(pooled, 0, (size_t)GG * CC * 4, stream);
  hipMemsetAsync(cnt,    0, (size_t)GG * 4, stream);
  k_pool <<<cdiv_i(NN * CC, B), B, 0, stream>>>(agg, batch, pooled, cnt);
  k_final<<<1, 32, 0, stream>>>(pooled, cnt, Wf, bf, out);
}
